// FanAutoEncoder_76278619177317
// MI455X (gfx1250) — matrix operand order repaired and checked by reading, not compiled, not run on hardware
//
#include <hip/hip_runtime.h>
#include <hip/hip_bf16.h>

// ---------------- types / helpers ----------------
typedef __attribute__((ext_vector_type(8)))  __bf16 v8bf;
typedef __attribute__((ext_vector_type(16))) __bf16 v16bf;
typedef __attribute__((ext_vector_type(8)))  float  v8f;

static __device__ inline __bf16 f2bf(float f) {
  unsigned u = __builtin_bit_cast(unsigned, f);
  unsigned r = u + 0x7FFFu + ((u >> 16) & 1u);   // round-to-nearest-even
  unsigned short h = (unsigned short)(r >> 16);
  return __builtin_bit_cast(__bf16, h);
}

static __device__ inline v8f vzero() {
  v8f z;
#pragma unroll
  for (int i = 0; i < 8; i++) z[i] = 0.0f;
  return z;
}

static __device__ inline v16bf make16(v8bf lo, v8bf hi) {
  v16bf r;
#pragma unroll
  for (int i = 0; i < 8; i++) { r[i] = lo[i]; r[i + 8] = hi[i]; }
  return r;
}

// A-fragment (16x32 bf16, row-major source, ldk elements per row).
// lanes 0-15: row=lane,   K = 0..7  and 16..23
// lanes 16-31: row=lane-16, K = 8..15 and 24..31
static __device__ inline v16bf load_afrag(const __bf16* base, int ldk) {
  int lane = threadIdx.x & 31;
  const __bf16* p = base + (size_t)(lane & 15) * ldk + ((lane & 16) ? 8 : 0);
  return make16(*(const v8bf*)p, *(const v8bf*)(p + 16));
}

// B-fragment (32x16 bf16) from W stored [N][K] row-major (i.e. B^T row-major).
// lanes 0-15: n=lane, K=0..7 and 16..23 ; lanes 16-31: n=lane-16, K=8..15 and 24..31
static __device__ inline v16bf load_bfrag(const __bf16* base, int ldk) {
  int lane = threadIdx.x & 31;
  const __bf16* p = base + (size_t)(lane & 15) * ldk + ((lane & 16) ? 8 : 0);
  return make16(*(const v8bf*)p, *(const v8bf*)(p + 16));
}

static __device__ inline v8f wmma_bf(v16bf a, v16bf b, v8f c) {
  return __builtin_amdgcn_wmma_f32_16x16x32_bf16(false, a, false, b, (short)0, c,
                                                 false, false);
}

// ---------------- constants ----------------
#define BB 64
#define TT 1024
#define DD 384
#define CC 64
#define HH 32
#define UU 128

// ---------------- kernel 1: pad + bf16-convert input ----------------
__global__ void pack_x_kernel(const float* __restrict__ x, __bf16* __restrict__ xpad) {
  size_t i = (size_t)blockIdx.x * blockDim.x + threadIdx.x;  // over B*(T+4)*D
  int d  = (int)(i % DD);
  size_t r = i / DD;
  int tp = (int)(r % (TT + 4));
  int b  = (int)(r / (TT + 4));
  float v = 0.0f;
  if (tp >= 2 && tp < TT + 2) v = x[((size_t)b * TT + (tp - 2)) * DD + d];
  xpad[i] = f2bf(v);
}

// ---------------- kernel 2: pack all weights to bf16 [N][K] layouts ----------
__global__ void pack_w_kernel(
    const float* conv_w, const float* enc_bb_w,
    const float* enc_ff1_w, const float* enc_ff2_w,
    const float* enc_ta_w, const float* enc_tb_w,
    const float* enc_ta_b, const float* enc_tb_b,
    const float* w_caps,  const float* dec_bb_w,
    const float* dec_ff1_w, const float* dec_ff2_w,
    const float* dec_ta_w, const float* dec_tb_w,
    const float* dec_ta_b, const float* dec_tb_b,
    const float* out_w,
    __bf16* wconv, __bf16* enc_wx, __bf16* enc_wh,
    __bf16* enc_w1, __bf16* enc_w2, __bf16* enc_wt, float* enc_tsb,
    __bf16* wcapsp, __bf16* dec_wx, __bf16* dec_wh,
    __bf16* dec_w1, __bf16* dec_w2, __bf16* dec_wt, float* dec_tsb,
    __bf16* woutp) {
  int tid = blockIdx.x * blockDim.x + threadIdx.x;
  int nth = gridDim.x * blockDim.x;
  // conv: [C=64][D=384][5] -> [5][64][384]
  for (int i = tid; i < 5 * CC * DD; i += nth) {
    int d = i % DD; int c = (i / DD) % CC; int k = i / (DD * CC);
    wconv[i] = f2bf(conv_w[((size_t)c * DD + d) * 5 + k]);
  }
  // enc backbone: [U=128][C+H=96] -> wx [128][64], wh [128][32]
  for (int i = tid; i < UU * CC; i += nth) {
    int k = i % CC; int u = i / CC;
    enc_wx[i] = f2bf(enc_bb_w[(size_t)u * (CC + HH) + k]);
  }
  for (int i = tid; i < UU * HH; i += nth) {
    int k = i % HH; int u = i / HH;
    enc_wh[i] = f2bf(enc_bb_w[(size_t)u * (CC + HH) + CC + k]);
  }
  // enc heads [32][128]
  for (int i = tid; i < HH * UU; i += nth) {
    enc_w1[i] = f2bf(enc_ff1_w[i]);
    enc_w2[i] = f2bf(enc_ff2_w[i]);
    enc_wt[i] = f2bf(enc_ta_w[i] + enc_tb_w[i]);
  }
  for (int i = tid; i < HH; i += nth) enc_tsb[i] = enc_ta_b[i] + enc_tb_b[i];
  // caps: [NC=4][H=32][CD=16] -> [64][32] with n = c*16+k
  for (int i = tid; i < 64 * HH; i += nth) {
    int h = i % HH; int n = i / HH; int c = n >> 4; int kq = n & 15;
    wcapsp[i] = f2bf(w_caps[((size_t)c * HH + h) * 16 + kq]);
  }
  // dec backbone: [128][64+32]
  for (int i = tid; i < UU * 64; i += nth) {
    int k = i % 64; int u = i / 64;
    dec_wx[i] = f2bf(dec_bb_w[(size_t)u * 96 + k]);
  }
  for (int i = tid; i < UU * HH; i += nth) {
    int k = i % HH; int u = i / HH;
    dec_wh[i] = f2bf(dec_bb_w[(size_t)u * 96 + 64 + k]);
  }
  for (int i = tid; i < HH * UU; i += nth) {
    dec_w1[i] = f2bf(dec_ff1_w[i]);
    dec_w2[i] = f2bf(dec_ff2_w[i]);
    dec_wt[i] = f2bf(dec_ta_w[i] + dec_tb_w[i]);
  }
  for (int i = tid; i < HH; i += nth) dec_tsb[i] = dec_ta_b[i] + dec_tb_b[i];
  // out projection [D=384][H=32]
  for (int i = tid; i < DD * HH; i += nth) woutp[i] = f2bf(out_w[i]);
}

// ---------------- kernel 3: conv1d(k=5) as WMMA GEMM + ReLU ----------------
// Each wave: one 16-row (t) tile x all 64 output channels.
// Tap weights are staged into LDS with async memory->LDS copies (ASYNCcnt).
__global__ void conv_gemm_kernel(const __bf16* __restrict__ xpad,   // [B][T+4][D]
                                 const __bf16* __restrict__ wpk,    // [5][64][384]
                                 const float* __restrict__ conv_b,
                                 __bf16* __restrict__ feat) {       // [B][T][64]
  __shared__ __align__(16) __bf16 s_w[CC * DD];  // one tap: 48KB
  int wid = threadIdx.x >> 5;
  int mtile = blockIdx.x * (blockDim.x >> 5) + wid;
  int tilesPerB = TT / 16;
  int b  = mtile / tilesPerB;
  int t0 = (mtile % tilesPerB) * 16;
  const __bf16* Abase = xpad + ((size_t)b * (TT + 4) + t0) * DD;

  v8f acc[4] = {vzero(), vzero(), vzero(), vzero()};
  for (int k = 0; k < 5; k++) {
    __syncthreads();  // everyone done reading previous tap
    {
      // async global -> LDS stage of this tap's 64x384 bf16 weights
      unsigned lds0 = (unsigned)(uintptr_t)s_w;
      const __bf16* g0 = wpk + (size_t)k * CC * DD;
      for (int i = (int)threadIdx.x * 8; i < CC * DD; i += (int)blockDim.x * 8) {
        unsigned loff = lds0 + (unsigned)(i * 2);
        unsigned long long ga = (unsigned long long)(uintptr_t)(g0 + i);
        asm volatile("global_load_async_to_lds_b128 %0, %1, off"
                     :: "v"(loff), "v"(ga) : "memory");
      }
      asm volatile("s_wait_asynccnt 0" ::: "memory");
    }
    __syncthreads();  // every wave waited its own ASYNCcnt -> LDS fully staged
    const __bf16* Ak = Abase + (size_t)k * DD;
    for (int kb = 0; kb < DD; kb += 32) {
      v16bf a = load_afrag(Ak + kb, DD);
      v16bf bw[4];
#pragma unroll
      for (int nb = 0; nb < 4; nb++)
        bw[nb] = load_bfrag(s_w + (size_t)(nb * 16) * DD + kb, DD);
#pragma unroll
      for (int nb = 0; nb < 4; nb++)
        acc[nb] = wmma_bf(a, bw[nb], acc[nb]);
    }
  }
  int lane = threadIdx.x & 31;
  int col0 = lane & 15;
  int rh = (lane & 16) ? 8 : 0;
#pragma unroll
  for (int nb = 0; nb < 4; nb++) {
    int c = nb * 16 + col0;
    float bias = conv_b[c];
#pragma unroll
    for (int v = 0; v < 8; v++) {
      float r = acc[nb][v] + bias;
      r = r > 0.0f ? r : 0.0f;
      feat[((size_t)b * TT + t0 + v + rh) * CC + c] = f2bf(r);
    }
  }
}

// ---------------- kernel 4: generic bf16 GEMM, f32 out (+bias) ----------------
// C[M,N] = A[M,K] * W[N,K]^T + bias.  Wave tile: 16M x 64N.
__global__ void gemm_bf16_kernel(const __bf16* __restrict__ A,
                                 const __bf16* __restrict__ W,
                                 const float* __restrict__ bias,
                                 float* __restrict__ Cout,
                                 int M, int N, int K) {
  int wid = threadIdx.x >> 5;
  int m0 = (blockIdx.x * (int)(blockDim.x >> 5) + wid) * 16;
  int n0 = blockIdx.y * 64;
  v8f acc[4] = {vzero(), vzero(), vzero(), vzero()};
  for (int kb = 0; kb < K; kb += 32) {
    v16bf a = load_afrag(A + (size_t)m0 * K + kb, K);
    v16bf bw[4];
#pragma unroll
    for (int nb = 0; nb < 4; nb++)
      bw[nb] = load_bfrag(W + (size_t)(n0 + nb * 16) * K + kb, K);
#pragma unroll
    for (int nb = 0; nb < 4; nb++)
      acc[nb] = wmma_bf(a, bw[nb], acc[nb]);
  }
  int lane = threadIdx.x & 31;
  int col0 = lane & 15;
  int rh = (lane & 16) ? 8 : 0;
#pragma unroll
  for (int nb = 0; nb < 4; nb++) {
    int n = n0 + nb * 16 + col0;
    float bv = bias[n];
#pragma unroll
    for (int v = 0; v < 8; v++)
      Cout[(size_t)(m0 + v + rh) * N + n] = acc[nb][v] + bv;
  }
}

// ---------------- kernel 5: CfC sequential scan (one wave / 16-batch tile) -----
// mode 0: encoder -> fused capsule projection + squash, writes caps f32 + bf16
// mode 1: decoder -> writes h (bf16) for the output projection GEMM
__global__ void cfc_scan_kernel(const float* __restrict__ xproj,   // [B][T][128]
                                const __bf16* __restrict__ wh,     // [128][32]
                                const __bf16* __restrict__ w1,     // [32][128]
                                const __bf16* __restrict__ w2,
                                const __bf16* __restrict__ wt,
                                const float* __restrict__ b1,
                                const float* __restrict__ b2,
                                const float* __restrict__ bt,
                                const __bf16* __restrict__ wcaps,  // [64][32]
                                float* __restrict__ caps_f,        // [B][T][64]
                                __bf16* __restrict__ caps_b,
                                __bf16* __restrict__ hout,         // [B][T][32]
                                int mode) {
  __shared__ __align__(16) __bf16 s_wh[UU * HH];
  __shared__ __align__(16) __bf16 s_w1[HH * UU];
  __shared__ __align__(16) __bf16 s_w2[HH * UU];
  __shared__ __align__(16) __bf16 s_wt[HH * UU];
  __shared__ __align__(16) __bf16 s_wc[64 * HH];
  __shared__ float s_b1[HH], s_b2[HH], s_bt[HH];
  __shared__ __align__(16) __bf16 s_h[16 * HH];
  __shared__ __align__(16) __bf16 s_z[16 * UU];

  int lane = threadIdx.x & 31;
  for (int i = lane; i < UU * HH; i += 32) s_wh[i] = wh[i];
  for (int i = lane; i < HH * UU; i += 32) {
    s_w1[i] = w1[i]; s_w2[i] = w2[i]; s_wt[i] = wt[i];
  }
  if (mode == 0)
    for (int i = lane; i < 64 * HH; i += 32) s_wc[i] = wcaps[i];
  if (lane < HH) { s_b1[lane] = b1[lane]; s_b2[lane] = b2[lane]; s_bt[lane] = bt[lane]; }
  for (int i = lane; i < 16 * HH; i += 32) s_h[i] = f2bf(0.0f);
  __syncthreads();

  int b0 = blockIdx.x * 16;
  int col0 = lane & 15;
  int rh = (lane & 16) ? 8 : 0;
  float hb1[2] = {s_b1[col0], s_b1[16 + col0]};
  float hb2[2] = {s_b2[col0], s_b2[16 + col0]};
  float hbt[2] = {s_bt[col0], s_bt[16 + col0]};

  for (int t = 0; t < TT; t++) {
    // ---- z = lecun_tanh(xproj + h @ wh^T) ----
    v8f zac[8];
#pragma unroll
    for (int nb = 0; nb < 8; nb++)
#pragma unroll
      for (int v = 0; v < 8; v++)
        zac[nb][v] = xproj[((size_t)(b0 + v + rh) * TT + t) * UU + nb * 16 + col0];
    if (t + 1 < TT)
      __builtin_prefetch(&xproj[((size_t)(b0 + col0) * TT + t + 1) * UU + (rh << 3)], 0, 0);

    v16bf ah = load_afrag(s_h, HH);
#pragma unroll
    for (int g = 0; g < 2; g++) {
      v16bf bw[4];
#pragma unroll
      for (int nb = 0; nb < 4; nb++)
        bw[nb] = load_bfrag(s_wh + (size_t)((g * 4 + nb) * 16) * HH, HH);
#pragma unroll
      for (int nb = 0; nb < 4; nb++)
        zac[g * 4 + nb] = wmma_bf(ah, bw[nb], zac[g * 4 + nb]);
    }
#pragma unroll
    for (int nb = 0; nb < 8; nb++)
#pragma unroll
      for (int v = 0; v < 8; v++) {
        float zz = 1.7159f * tanhf(0.666f * zac[nb][v]);
        s_z[(v + rh) * UU + nb * 16 + col0] = f2bf(zz);
      }
    __syncthreads();

    // ---- heads: ff1, ff2, (ta+tb) ----
    v8f a1[2], a2[2], at[2];
#pragma unroll
    for (int nbh = 0; nbh < 2; nbh++)
#pragma unroll
      for (int v = 0; v < 8; v++) {
        a1[nbh][v] = hb1[nbh]; a2[nbh][v] = hb2[nbh]; at[nbh][v] = hbt[nbh];
      }
#pragma unroll
    for (int kk = 0; kk < 4; kk++) {
      v16bf az = load_afrag(s_z + kk * 32, UU);
      v16bf bw1[2], bw2[2], bwt[2];
#pragma unroll
      for (int nbh = 0; nbh < 2; nbh++) {
        bw1[nbh] = load_bfrag(s_w1 + (size_t)(nbh * 16) * UU + kk * 32, UU);
        bw2[nbh] = load_bfrag(s_w2 + (size_t)(nbh * 16) * UU + kk * 32, UU);
        bwt[nbh] = load_bfrag(s_wt + (size_t)(nbh * 16) * UU + kk * 32, UU);
      }
#pragma unroll
      for (int nbh = 0; nbh < 2; nbh++) {
        a1[nbh] = wmma_bf(az, bw1[nbh], a1[nbh]);
        a2[nbh] = wmma_bf(az, bw2[nbh], a2[nbh]);
        at[nbh] = wmma_bf(az, bwt[nbh], at[nbh]);
      }
    }
    // ---- gate + new hidden ----
#pragma unroll
    for (int nbh = 0; nbh < 2; nbh++)
#pragma unroll
      for (int v = 0; v < 8; v++) {
        float f1 = tanhf(a1[nbh][v]);
        float f2 = tanhf(a2[nbh][v]);
        float ti = 1.0f / (1.0f + __expf(-at[nbh][v]));
        float hn = f1 * (1.0f - ti) + ti * f2;
        s_h[(v + rh) * HH + nbh * 16 + col0] = f2bf(hn);
        if (mode == 1)
          hout[((size_t)(b0 + v + rh) * TT + t) * HH + nbh * 16 + col0] = f2bf(hn);
      }
    __syncthreads();

    if (mode == 0) {
      // ---- capsules: caps = squash(h @ wcaps^T), per-capsule dim = 16 lanes ----
      v16bf ahn = load_afrag(s_h, HH);
      v16bf bwc[4];
#pragma unroll
      for (int nc = 0; nc < 4; nc++)
        bwc[nc] = load_bfrag(s_wc + (size_t)(nc * 16) * HH, HH);
      v8f cc[4];
#pragma unroll
      for (int nc = 0; nc < 4; nc++)
        cc[nc] = wmma_bf(ahn, bwc[nc], vzero());
#pragma unroll
      for (int nc = 0; nc < 4; nc++) {
#pragma unroll
        for (int v = 0; v < 8; v++) {
          float val = cc[nc][v];
          float m = val * val;
          m += __shfl_xor(m, 1);
          m += __shfl_xor(m, 2);
          m += __shfl_xor(m, 4);
          m += __shfl_xor(m, 8);
          float sc = m / (1.0f + m) * rsqrtf(m + 1e-8f);
          float o = val * sc;
          size_t oi = ((size_t)(b0 + v + rh) * TT + t) * 64 + nc * 16 + col0;
          caps_f[oi] = o;
          caps_b[oi] = f2bf(o);
        }
      }
    }
  }
}

// ---------------- host side ----------------
extern "C" void kernel_launch(void* const* d_in, const int* in_sizes, int n_in,
                              void* d_out, int out_size, void* d_ws, size_t ws_size,
                              hipStream_t stream) {
  const float* x        = (const float*)d_in[0];
  const float* conv_w   = (const float*)d_in[1];
  const float* conv_b   = (const float*)d_in[2];
  const float* enc_bb_w = (const float*)d_in[3];
  const float* enc_bb_b = (const float*)d_in[4];
  const float* enc_ff1w = (const float*)d_in[5];
  const float* enc_ff1b = (const float*)d_in[6];
  const float* enc_ff2w = (const float*)d_in[7];
  const float* enc_ff2b = (const float*)d_in[8];
  const float* enc_taw  = (const float*)d_in[9];
  const float* enc_tab  = (const float*)d_in[10];
  const float* enc_tbw  = (const float*)d_in[11];
  const float* enc_tbb  = (const float*)d_in[12];
  const float* w_caps   = (const float*)d_in[13];
  const float* dec_bb_w = (const float*)d_in[14];
  const float* dec_bb_b = (const float*)d_in[15];
  const float* dec_ff1w = (const float*)d_in[16];
  const float* dec_ff1b = (const float*)d_in[17];
  const float* dec_ff2w = (const float*)d_in[18];
  const float* dec_ff2b = (const float*)d_in[19];
  const float* dec_taw  = (const float*)d_in[20];
  const float* dec_tab  = (const float*)d_in[21];
  const float* dec_tbw  = (const float*)d_in[22];
  const float* dec_tbb  = (const float*)d_in[23];
  const float* out_w    = (const float*)d_in[24];
  const float* out_b    = (const float*)d_in[25];

  const size_t BT = (size_t)BB * TT;
  unsigned char* ws = (unsigned char*)d_ws;
  size_t off = 0;
  auto take = [&](size_t bytes) -> void* {
    void* p = ws + off;
    off += (bytes + 255) & ~(size_t)255;
    return p;
  };
  __bf16* xpad    = (__bf16*)take((size_t)BB * (TT + 4) * DD * 2);
  __bf16* feat    = (__bf16*)take(BT * CC * 2);
  float*  xproj   = (float*)take(BT * UU * 4);
  __bf16* caps_bf = (__bf16*)take(BT * 64 * 2);
  __bf16* hdec    = (__bf16*)take(BT * HH * 2);
  __bf16* wconv   = (__bf16*)take((size_t)5 * CC * DD * 2);
  __bf16* enc_wx  = (__bf16*)take((size_t)UU * CC * 2);
  __bf16* enc_wh  = (__bf16*)take((size_t)UU * HH * 2);
  __bf16* enc_w1  = (__bf16*)take((size_t)HH * UU * 2);
  __bf16* enc_w2  = (__bf16*)take((size_t)HH * UU * 2);
  __bf16* enc_wt  = (__bf16*)take((size_t)HH * UU * 2);
  float*  enc_tsb = (float*)take(HH * 4);
  __bf16* wcapsp  = (__bf16*)take((size_t)64 * HH * 2);
  __bf16* dec_wx  = (__bf16*)take((size_t)UU * 64 * 2);
  __bf16* dec_wh  = (__bf16*)take((size_t)UU * HH * 2);
  __bf16* dec_w1  = (__bf16*)take((size_t)HH * UU * 2);
  __bf16* dec_w2  = (__bf16*)take((size_t)HH * UU * 2);
  __bf16* dec_wt  = (__bf16*)take((size_t)HH * UU * 2);
  float*  dec_tsb = (float*)take(HH * 4);
  __bf16* woutp   = (__bf16*)take((size_t)DD * HH * 2);

  float* recon  = (float*)d_out;
  float* caps_f = (float*)d_out + BT * DD;

  // 1) pack input (pad + bf16)
  {
    size_t total = (size_t)BB * (TT + 4) * DD;
    pack_x_kernel<<<dim3((unsigned)(total / 256)), 256, 0, stream>>>(x, xpad);
  }
  // 2) pack weights
  pack_w_kernel<<<64, 256, 0, stream>>>(
      conv_w, enc_bb_w, enc_ff1w, enc_ff2w, enc_taw, enc_tbw, enc_tab, enc_tbb,
      w_caps, dec_bb_w, dec_ff1w, dec_ff2w, dec_taw, dec_tbw, dec_tab, dec_tbb,
      out_w,
      wconv, enc_wx, enc_wh, enc_w1, enc_w2, enc_wt, enc_tsb, wcapsp,
      dec_wx, dec_wh, dec_w1, dec_w2, dec_wt, dec_tsb, woutp);
  // 3) conv + relu -> feat (bf16)
  conv_gemm_kernel<<<dim3((unsigned)(BT / 16 / 4)), 128, 0, stream>>>(
      xpad, wconv, conv_b, feat);
  // 4) enc x_proj = feat @ wx^T + b  -> [B,T,128] f32
  gemm_bf16_kernel<<<dim3((unsigned)(BT / 64), UU / 64), 128, 0, stream>>>(
      feat, enc_wx, enc_bb_b, xproj, (int)BT, UU, CC);
  // 5) encoder scan (+ fused capsules/squash)
  cfc_scan_kernel<<<dim3(BB / 16), 32, 0, stream>>>(
      xproj, enc_wh, enc_w1, enc_w2, enc_wt, enc_ff1b, enc_ff2b, enc_tsb,
      wcapsp, caps_f, caps_bf, (__bf16*)nullptr, 0);
  // 6) dec x_proj = caps @ wx^T + b -> [B,T,128] f32 (reuse buffer)
  gemm_bf16_kernel<<<dim3((unsigned)(BT / 64), UU / 64), 128, 0, stream>>>(
      caps_bf, dec_wx, dec_bb_b, xproj, (int)BT, UU, 64);
  // 7) decoder scan -> hdec (bf16)
  cfc_scan_kernel<<<dim3(BB / 16), 32, 0, stream>>>(
      xproj, dec_wh, dec_w1, dec_w2, dec_wt, dec_ff1b, dec_ff2b, dec_tsb,
      (__bf16*)nullptr, (float*)nullptr, (__bf16*)nullptr, hdec, 1);
  // 8) recon = hdec @ out_w^T + out_b -> [B,T,384] f32
  gemm_bf16_kernel<<<dim3((unsigned)(BT / 64), DD / 64), 128, 0, stream>>>(
      hdec, woutp, out_b, recon, (int)BT, DD, HH);
  (void)in_sizes; (void)n_in; (void)out_size; (void)ws_size;
}
